// StabilityGNN_51857435132133
// MI455X (gfx1250) — matrix operand order repaired and checked by reading, not compiled, not run on hardware
//
#include <hip/hip_runtime.h>
#include <hip/hip_bf16.h>

#define FEATS 64
#define NCLS 4
#define TOPOD 16
#define NGRAPH 64

typedef __attribute__((ext_vector_type(16))) __bf16 v16bf;
typedef __attribute__((ext_vector_type(8)))  float  v8f;

union FragB16 { v16bf v; unsigned short s[16]; uint4 q[2]; };
union FragF32 { v8f v; float f[8]; };

__device__ __forceinline__ unsigned short f32_to_bf16(float x) {
    unsigned u = __float_as_uint(x);
    return (unsigned short)((u + 0x7FFFu + ((u >> 16) & 1u)) >> 16);  // RNE
}

// ---------------- elementwise / setup kernels ----------------

__global__ void k_zero(float* __restrict__ p, int n) {
    int i = blockIdx.x * blockDim.x + threadIdx.x;
    if (i < n) p[i] = 0.0f;
}

__global__ void k_degree(const int* __restrict__ dst, float* __restrict__ deg, int E) {
    int e = blockIdx.x * blockDim.x + threadIdx.x;
    if (e < E) atomicAdd(&deg[dst[e]], 1.0f);
}

__global__ void k_deg_inv_sqrt(float* __restrict__ deg, int N) {
    int i = blockIdx.x * blockDim.x + threadIdx.x;
    if (i < N) deg[i] = rsqrtf(deg[i] + 1.0f);  // +1 for self-loop
}

// ---------------- bf16 WMMA GEMM: C[N,64] = bf16(A[N,64]) @ bf16(W[64,64]) ----------------
// f32 inputs converted in-kernel. W^T staged in LDS (bf16) so B fragments are 2x ds_load_b128.
// One wave per 16-row tile (grid-stride); 4 N-tiles x 2 K-steps = 8 v_wmma per tile.
// Full tiles take an unguarded store path (base + immediate offsets).

__global__ void __launch_bounds__(256)
k_gemm_bf16(const float* __restrict__ A, const float* __restrict__ W,
            float* __restrict__ C, int nrows) {
    __shared__ __align__(16) unsigned short lwT[FEATS * FEATS];  // W transposed, bf16 (8 KB)

    // cooperative transpose+convert of W into LDS
    for (int i = threadIdx.x; i < FEATS * FEATS; i += 256) {
        int r = i >> 6, c = i & 63;                 // coalesced global read of W[r][c]
        lwT[c * FEATS + r] = f32_to_bf16(W[i]);
    }
    __syncthreads();

    const int lane = threadIdx.x & 31;
    const int wid  = threadIdx.x >> 5;
    const int col  = lane & 15;
    const int hi   = lane >> 4;   // lane half selects K sub-range per ISA layout

    // B fragments (32x16 bf16): V0..7 lanes0-15 hold K=kb..kb+15, lanes16-31 K=kb+16..kb+31.
    // With W^T in LDS those 16 K values are contiguous -> two b128 LDS loads per fragment.
    FragB16 bf[2][4];
#pragma unroll
    for (int kt = 0; kt < 2; ++kt) {
        int kbase = kt * 32 + hi * 16;
#pragma unroll
        for (int nt = 0; nt < 4; ++nt) {
            const uint4* p = (const uint4*)&lwT[(nt * 16 + col) * FEATS + kbase];
            bf[kt][nt].q[0] = p[0];
            bf[kt][nt].q[1] = p[1];
        }
    }

    const int ntiles = (nrows + 15) >> 4;
    const int stride = (int)gridDim.x * 8;
    for (int tile = (int)blockIdx.x * 8 + wid; tile < ntiles; tile += stride) {
        const int rt = tile * 16;

        // A fragments (16x32 bf16): lanes0-15 K [0..15], lanes16-31 K [16..31]
        int row = rt + col;
        if (row >= nrows) row = nrows - 1;
        const float* arow = A + (size_t)row * FEATS;
        FragB16 af[2];
#pragma unroll
        for (int kt = 0; kt < 2; ++kt) {
            int k0 = kt * 32 + hi * 16;
            int k1 = k0 + 8;
            float4 f0 = *(const float4*)(arow + k0);
            float4 f1 = *(const float4*)(arow + k0 + 4);
            float4 f2 = *(const float4*)(arow + k1);
            float4 f3 = *(const float4*)(arow + k1 + 4);
            const float tmp[16] = {f0.x, f0.y, f0.z, f0.w, f1.x, f1.y, f1.z, f1.w,
                                   f2.x, f2.y, f2.z, f2.w, f3.x, f3.y, f3.z, f3.w};
#pragma unroll
            for (int j = 0; j < 16; ++j) af[kt].s[j] = f32_to_bf16(tmp[j]);
        }

        FragF32 acc[4];
#pragma unroll
        for (int nt = 0; nt < 4; ++nt) {
            v8f c = {};
            c = __builtin_amdgcn_wmma_f32_16x16x32_bf16(false, af[0].v, false, bf[0][nt].v,
                                                        (short)0, c, false, false);
            c = __builtin_amdgcn_wmma_f32_16x16x32_bf16(false, af[1].v, false, bf[1][nt].v,
                                                        (short)0, c, false, false);
            acc[nt].v = c;
        }

        // C/D layout: VGPR r -> row rt + r + hi*8, col = nt*16 + (lane&15)
        if (rt + 16 <= nrows) {
            // full tile: one base address, all offsets compile-time immediates
            float* cbase = C + (size_t)(rt + hi * 8) * FEATS + col;
#pragma unroll
            for (int nt = 0; nt < 4; ++nt)
#pragma unroll
                for (int r = 0; r < 8; ++r)
                    cbase[r * FEATS + nt * 16] = acc[nt].f[r];
        } else {
            // tail tile (unused when N % 16 == 0)
#pragma unroll
            for (int nt = 0; nt < 4; ++nt)
#pragma unroll
                for (int r = 0; r < 8; ++r) {
                    int ro = rt + r + hi * 8;
                    if (ro < nrows) C[(size_t)ro * FEATS + nt * 16 + col] = acc[nt].f[r];
                }
        }
    }
}

// ---------------- edge scatter: one wave per edge, 2 feats per lane ----------------

__global__ void k_edge_msg(const int* __restrict__ src, const int* __restrict__ dst,
                           const float* __restrict__ dis, const float* __restrict__ xw,
                           float* __restrict__ agg, int E) {
    int gid = blockIdx.x * blockDim.x + threadIdx.x;   // E*32 = 51.2M < 2^31
    int e = gid >> 5;
    if (e >= E) return;
    int lane = gid & 31;
    int s = src[e], d = dst[e];
    float nrm = dis[s] * dis[d];
    const float2* xs = (const float2*)(xw + (size_t)s * FEATS);
    __builtin_prefetch(agg + (size_t)d * FEATS, 0, 1);   // global_prefetch of scatter row
    float2 v = xs[lane];
    float* ap = agg + (size_t)d * FEATS + lane * 2;
    atomicAdd(ap,     v.x * nrm);   // L2-resident f32 atomics (xw table is 25.6MB << 192MB L2)
    atomicAdd(ap + 1, v.y * nrm);
}

// ---------------- self-loop + bias + ReLU (in place on agg) ----------------

__global__ void k_selfloop_relu(float* __restrict__ agg, const float* __restrict__ xw,
                                const float* __restrict__ dis, const float* __restrict__ bias,
                                int NF) {
    int t = blockIdx.x * blockDim.x + threadIdx.x;
    if (t >= NF) return;
    int i = t >> 6;
    int f = t & 63;
    float di = dis[i];
    float v = agg[t] + xw[t] * di * di + bias[f];
    agg[t] = v > 0.0f ? v : 0.0f;
}

// ---------------- fused layer-2 epilogue: self-loop+bias+ReLU -> mean-pool atomics ----------------
// h2 is never materialized; its only consumer is the pool.

__global__ void k_selfloop_relu_pool(const float* __restrict__ agg, const float* __restrict__ xw,
                                     const float* __restrict__ dis, const float* __restrict__ bias,
                                     const int* __restrict__ batch,
                                     float* __restrict__ sums, float* __restrict__ cnt,
                                     int NF) {
    int t = blockIdx.x * blockDim.x + threadIdx.x;
    if (t >= NF) return;
    int i = t >> 6;
    int f = t & 63;
    float di = dis[i];
    float v = agg[t] + xw[t] * di * di + bias[f];
    v = v > 0.0f ? v : 0.0f;
    int g = batch[i];
    atomicAdd(&sums[g * FEATS + f], v);
    if (f == 0) atomicAdd(&cnt[g], 1.0f);
}

// ---------------- final head: [64, 64+16] @ [80, 4] + b ----------------

__global__ void k_final(const float* __restrict__ sums, const float* __restrict__ cnt,
                        const float* __restrict__ topo, const float* __restrict__ Wlin,
                        const float* __restrict__ blin, float* __restrict__ out) {
    int t = threadIdx.x;
    if (t >= NGRAPH * NCLS) return;
    int g = t >> 2, c = t & 3;
    float inv = 1.0f / fmaxf(cnt[g], 1.0f);
    float acc = blin[c];
    for (int f = 0; f < FEATS; ++f)
        acc += sums[g * FEATS + f] * inv * Wlin[f * NCLS + c];
    for (int k = 0; k < TOPOD; ++k)
        acc += topo[g * TOPOD + k] * Wlin[(FEATS + k) * NCLS + c];
    out[g * NCLS + c] = acc;
}

// ---------------- host orchestration ----------------

static inline unsigned blks(long long n, int b) { return (unsigned)((n + b - 1) / b); }

extern "C" void kernel_launch(void* const* d_in, const int* in_sizes, int n_in,
                              void* d_out, int out_size, void* d_ws, size_t ws_size,
                              hipStream_t stream) {
    const float* x    = (const float*)d_in[0];
    const int*   ei   = (const int*)d_in[1];
    const int*   batch= (const int*)d_in[2];
    const float* topo = (const float*)d_in[3];
    const float* W1   = (const float*)d_in[4];
    const float* b1   = (const float*)d_in[5];
    const float* W2   = (const float*)d_in[6];
    const float* b2   = (const float*)d_in[7];
    const float* Wlin = (const float*)d_in[8];
    const float* blin = (const float*)d_in[9];
    float* out = (float*)d_out;

    int N = in_sizes[0] / FEATS;   // 100000
    int E = in_sizes[1] / 2;       // 1600000
    const int* src = ei;
    const int* dst = ei + E;

    // workspace carve-up (all f32; ~51.6 MB)
    float* ws   = (float*)d_ws;
    float* dis  = ws;                          // N      (degree -> deg_inv_sqrt, in place)
    float* xw   = dis + N;                     // N*64   (x@W per layer)
    float* agg  = xw + (size_t)N * FEATS;      // N*64   (aggregation, reused as h1)
    float* sums = agg + (size_t)N * FEATS;     // 64*64
    float* cnt  = sums + NGRAPH * FEATS;       // 64

    const int B = 256;
    int NF = N * FEATS;
    long long edge_threads = (long long)E * 32;
    // GEMM: 8 waves/block, ~4 row-tiles per wave via grid-stride
    unsigned gemm_blocks = 200;

    // degree -> deg_inv_sqrt
    k_zero<<<blks(N, B), B, 0, stream>>>(dis, N);
    k_degree<<<blks(E, B), B, 0, stream>>>(dst, dis, E);
    k_deg_inv_sqrt<<<blks(N, B), B, 0, stream>>>(dis, N);

    // ----- layer 1 -----
    k_zero<<<blks(NF, B), B, 0, stream>>>(agg, NF);
    k_gemm_bf16<<<gemm_blocks, B, 0, stream>>>(x, W1, xw, N);
    k_edge_msg<<<blks(edge_threads, B), B, 0, stream>>>(src, dst, dis, xw, agg, E);
    k_selfloop_relu<<<blks(NF, B), B, 0, stream>>>(agg, xw, dis, b1, NF);   // h1 in agg

    // ----- layer 2 -----
    k_gemm_bf16<<<gemm_blocks, B, 0, stream>>>(agg, W2, xw, N);             // h1@W2
    k_zero<<<blks(NF, B), B, 0, stream>>>(agg, NF);                         // h1 dead after GEMM
    k_edge_msg<<<blks(edge_threads, B), B, 0, stream>>>(src, dst, dis, xw, agg, E);

    // ----- fused epilogue + pool + head -----
    k_zero<<<blks(NGRAPH * FEATS + NGRAPH, B), B, 0, stream>>>(sums, NGRAPH * FEATS + NGRAPH);
    k_selfloop_relu_pool<<<blks(NF, B), B, 0, stream>>>(agg, xw, dis, b2, batch, sums, cnt, NF);
    k_final<<<1, 256, 0, stream>>>(sums, cnt, topo, Wlin, blin, out);
}
